// Experts_69647189672119
// MI455X (gfx1250) — matrix operand order repaired and checked by reading, not compiled, not run on hardware
//
#include <hip/hip_runtime.h>
#include <math.h>

// ---------------- problem constants (match reference) ----------------
#define NN   30000
#define EE   480000
#define HHX  128
#define LLX  3
#define KKX  4
#define CCX  10
#define GGX  600
#define NF   (NN * HHX)

typedef __bf16 bf16t;
typedef __attribute__((ext_vector_type(16))) __bf16 v16bf;
typedef __attribute__((ext_vector_type(8)))  float  v8f;
typedef __attribute__((ext_vector_type(4)))  unsigned int v4u;

union FragB { v16bf v; v4u q[2]; };

static __device__ __forceinline__ v8f wmma_bf16(const v16bf& a, const v16bf& b, const v8f& c) {
  return __builtin_amdgcn_wmma_f32_16x16x32_bf16(false, a, false, b, (short)0, c, false, false);
}

// async 16B global->LDS copy (CDNA5 VGLOBAL async, GVS mode, tracked by ASYNCcnt)
static __device__ __forceinline__ void async_copy16(void* lds_dst, const void* saddr_base,
                                                    unsigned byte_off) {
  unsigned lo = (unsigned)(size_t)lds_dst;
  asm volatile("global_load_async_to_lds_b128 %0, %1, %2"
               :: "v"(lo), "v"(byte_off), "s"(saddr_base)
               : "memory");
}
static __device__ __forceinline__ void async_wait() {
  asm volatile("s_wait_asynccnt 0x0" ::: "memory");
}

// ---------------- fp32 -> bf16 transposed weight conversion ----------------
// d[m][n][k] = (bf16) s[m][k][n]   (stores W^T so LDS staging is a straight copy)
__global__ void k_convT_bf16(const float* __restrict__ s, bf16t* __restrict__ d,
                             int K, int N, int total) {
  int i = blockIdx.x * blockDim.x + threadIdx.x;
  if (i >= total) return;
  int per = K * N;
  int m = i / per, r = i - m * per;
  int n = r / K, k = r - n * K;
  d[i] = (bf16t)s[(size_t)m * per + (size_t)k * N + n];
}

// ---------------- edge gather + scatter-add (segment_sum) ----------------
__global__ void k_gather_scatter(const float* __restrict__ h,
                                 const int* __restrict__ ei,     // [2,E]
                                 const float* __restrict__ ew,   // may be null
                                 float* __restrict__ agg, int E) {
  int t = blockIdx.x * blockDim.x + threadIdx.x;
  int e = t >> 5;
  if (e >= E) return;
  int lane = t & 31;
  int s = ei[e];
  int d = ei[E + e];
  // warm L2/L0 for a future edge's source row (global_prefetch_b8)
  int ep = e + 2048;
  if (ep < E) __builtin_prefetch(h + (size_t)ei[ep] * HHX + lane * 4, 0, 0);
  float w = ew ? ew[e] : 1.0f;
  const float4 v = *(const float4*)(h + (size_t)s * HHX + lane * 4);
  float* ap = agg + (size_t)d * HHX + lane * 4;
  atomicAdd(ap + 0, v.x * w);
  atomicAdd(ap + 1, v.y * w);
  atomicAdd(ap + 2, v.z * w);
  atomicAdd(ap + 3, v.w * w);
}

// ---------------- fused GIN layer: ((1+eps)h + agg) -> relu(W1) -> relu(W2) ----------------
// 64-row tile, 128 threads = 4 waves; weights arrive pre-transposed [n][k] via async DMA.
__launch_bounds__(128)
__global__ void k_gin_update_wmma(const float* __restrict__ hin,
                                  const float* __restrict__ agg,
                                  const float* __restrict__ epsp, int epsIdx,
                                  const bf16t* __restrict__ W1t, const float* __restrict__ b1,
                                  const bf16t* __restrict__ W2t, const float* __restrict__ b2,
                                  float* __restrict__ hout,
                                  bf16t* __restrict__ houtbf, int nrows) {
  __shared__ __align__(16) bf16t sA[64 * HHX];     // 16 KB activations (bf16)
  __shared__ __align__(16) bf16t sB[HHX * HHX];    // 32 KB weight (already [n][k])
  const int tid  = threadIdx.x;
  const int row0 = blockIdx.x * 64;
  const float epsv = 1.0f + epsp[epsIdx];

  // async-stage W1^T into LDS (contiguous copy), overlapped with A-tile compute
  for (int idx = tid; idx < (HHX * HHX) / 8; idx += 128)
    async_copy16(sB + idx * 8, W1t, (unsigned)(idx * 16));

  // A tile: (1+eps)*h + agg, converted to bf16
  for (int idx = tid; idx < 64 * HHX; idx += 128) {
    int r = idx >> 7, c = idx & 127;
    int row = row0 + r;
    float v = 0.0f;
    if (row < nrows) v = epsv * hin[(size_t)row * HHX + c] + agg[(size_t)row * HHX + c];
    sA[idx] = (bf16t)v;
  }
  async_wait();
  __syncthreads();

  const int wave = tid >> 5, lane = tid & 31;
  const int m0   = wave * 16;
  const int arow = m0 + (lane & 15);
  const int bcol = lane & 15;
  const int hi   = (lane >> 4) & 1;

  v8f acc[8];
  const v8f vzero = {0.f, 0.f, 0.f, 0.f, 0.f, 0.f, 0.f, 0.f};
#pragma unroll
  for (int t = 0; t < 8; ++t) acc[t] = vzero;

  // ---- GEMM1 ----
#pragma unroll
  for (int ks = 0; ks < 4; ++ks) {
    const int k0 = ks * 32;
    FragB a;
    const int ka = k0 + (hi ? 8 : 0);
    const bf16t* ap = &sA[arow * HHX + ka];
    a.q[0] = *(const v4u*)(ap);
    a.q[1] = *(const v4u*)(ap + 16);
#pragma unroll
    for (int t = 0; t < 8; ++t) {
      FragB b;
      const int kb = k0 + (hi ? 8 : 0);
      const bf16t* bp = &sB[((t * 16) + bcol) * HHX + kb];
      b.q[0] = *(const v4u*)(bp);
      b.q[1] = *(const v4u*)(bp + 16);
      acc[t] = wmma_bf16(a.v, b.v, acc[t]);
    }
  }
  // bias1 + relu -> back to sA (wave-private rows)
#pragma unroll
  for (int t = 0; t < 8; ++t) {
    const int col = t * 16 + bcol;
    const float bb = b1[col];
#pragma unroll
    for (int r = 0; r < 8; ++r) {
      const int rl = m0 + r + (hi ? 8 : 0);
      float v = acc[t][r] + bb;
      sA[rl * HHX + col] = (bf16t)(v > 0.f ? v : 0.f);
    }
  }
  __syncthreads();
  // async-stage W2^T
  for (int idx = tid; idx < (HHX * HHX) / 8; idx += 128)
    async_copy16(sB + idx * 8, W2t, (unsigned)(idx * 16));
  async_wait();
  __syncthreads();

#pragma unroll
  for (int t = 0; t < 8; ++t) acc[t] = vzero;

  // ---- GEMM2 ----
#pragma unroll
  for (int ks = 0; ks < 4; ++ks) {
    const int k0 = ks * 32;
    FragB a;
    const int ka = k0 + (hi ? 8 : 0);
    const bf16t* ap = &sA[arow * HHX + ka];
    a.q[0] = *(const v4u*)(ap);
    a.q[1] = *(const v4u*)(ap + 16);
#pragma unroll
    for (int t = 0; t < 8; ++t) {
      FragB b;
      const int kb = k0 + (hi ? 8 : 0);
      const bf16t* bp = &sB[((t * 16) + bcol) * HHX + kb];
      b.q[0] = *(const v4u*)(bp);
      b.q[1] = *(const v4u*)(bp + 16);
      acc[t] = wmma_bf16(a.v, b.v, acc[t]);
    }
  }
  // bias2 + relu -> global (fp32, optional bf16 mirror)
#pragma unroll
  for (int t = 0; t < 8; ++t) {
    const int col = t * 16 + bcol;
    const float bb = b2[col];
#pragma unroll
    for (int r = 0; r < 8; ++r) {
      const int row = row0 + m0 + r + (hi ? 8 : 0);
      if (row < nrows) {
        float v = acc[t][r] + bb;
        v = v > 0.f ? v : 0.f;
        hout[(size_t)row * HHX + col] = v;
        if (houtbf) houtbf[(size_t)row * HHX + col] = (bf16t)v;
      }
    }
  }
}

// ---------------- per-expert edge-mask MLP + hard-concrete gate ----------------
// 64 edges/block, 128 threads = 4 waves. edge_feat = [Z[src], Z[dst]] (K=256),
// gathered directly into LDS with async DMA.
__launch_bounds__(128)
__global__ void k_edge_mask_wmma(const bf16t* __restrict__ Zbf,
                                 const int* __restrict__ ei,
                                 const bf16t* __restrict__ W1t,  // [128][256] bf16 (W^T)
                                 const float* __restrict__ b1,   // [128]
                                 const float* __restrict__ w2,   // [128]
                                 const float* __restrict__ b2s,  // [1]
                                 float* __restrict__ eon,
                                 float* __restrict__ nw,
                                 float* __restrict__ out_em,     // edge_masks [E,K]
                                 int kExp, int E) {
  __shared__ __align__(16) bf16t sE[64 * 256];   // 32 KB, later aliased as hidden f32
  __shared__ __align__(16) bf16t sB[HHX * 64];   // 16 KB, W^T k-chunk [n][64]
  __shared__ float sw2[HHX];
  __shared__ int sSrc[64], sDst[64];
  const int tid = threadIdx.x;
  const int e0  = blockIdx.x * 64;

  if (tid < HHX) sw2[tid] = w2[tid];
  if (tid < 64) {
    int e = e0 + tid;
    sSrc[tid] = (e < E) ? ei[e] : 0;
    sDst[tid] = (e < E) ? ei[E + e] : 0;
  }
  __syncthreads();

  // async-gather edge features: 32 x 16B chunks per edge row, per-lane LDS dest
  for (int idx = tid; idx < 64 * 32; idx += 128) {
    const int r = idx >> 5;
    const int c = idx & 31;
    const int off = c * 8;   // bf16 offset within 256-wide row
    const unsigned gofs = (c < 16)
        ? (unsigned)(((size_t)sSrc[r] * HHX + off) * 2)
        : (unsigned)(((size_t)sDst[r] * HHX + (off - 128)) * 2);
    async_copy16(sE + r * 256 + off, Zbf, gofs);
  }
  // async-stage first W^T chunk (k = 0..63 for every n)
  for (int idx = tid; idx < 1024; idx += 128) {
    const int n = idx >> 3, c = idx & 7;
    async_copy16(sB + n * 64 + c * 8, W1t, (unsigned)((n * 256 + c * 8) * 2));
  }
  async_wait();
  __syncthreads();

  const int wave = tid >> 5, lane = tid & 31;
  const int m0   = wave * 16;
  const int arow = m0 + (lane & 15);
  const int bcol = lane & 15;
  const int hi   = (lane >> 4) & 1;

  v8f acc[8];
  const v8f vzero = {0.f, 0.f, 0.f, 0.f, 0.f, 0.f, 0.f, 0.f};
#pragma unroll
  for (int t = 0; t < 8; ++t) acc[t] = vzero;

  for (int kc = 0; kc < 4; ++kc) {          // K = 256 in chunks of 64
#pragma unroll
    for (int ks = 0; ks < 2; ++ks) {
      const int k0 = ks * 32;
      FragB a;
      const int ka = kc * 64 + k0 + (hi ? 8 : 0);
      const bf16t* ap = &sE[arow * 256 + ka];
      a.q[0] = *(const v4u*)(ap);
      a.q[1] = *(const v4u*)(ap + 16);
#pragma unroll
      for (int t = 0; t < 8; ++t) {
        FragB b;
        const int kb = k0 + (hi ? 8 : 0);
        const bf16t* bp = &sB[((t * 16) + bcol) * 64 + kb];
        b.q[0] = *(const v4u*)(bp);
        b.q[1] = *(const v4u*)(bp + 16);
        acc[t] = wmma_bf16(a.v, b.v, acc[t]);
      }
    }
    __syncthreads();
    if (kc < 3) {   // async-stage next W^T chunk
      for (int idx = tid; idx < 1024; idx += 128) {
        const int n = idx >> 3, c = idx & 7;
        async_copy16(sB + n * 64 + c * 8,
                     W1t, (unsigned)((n * 256 + (kc + 1) * 64 + c * 8) * 2));
      }
      async_wait();
      __syncthreads();
    }
  }

  // bias + relu -> hidden (alias LDS as float [64][128])
  float* hid = (float*)sE;
#pragma unroll
  for (int t = 0; t < 8; ++t) {
    const int col = t * 16 + bcol;
    const float bb = b1[col];
#pragma unroll
    for (int r = 0; r < 8; ++r) {
      const int rl = m0 + r + (hi ? 8 : 0);
      float v = acc[t][r] + bb;
      hid[rl * HHX + col] = v > 0.f ? v : 0.f;
    }
  }
  __syncthreads();

  // eml = hidden . w2 + b2 ; two threads per edge, 64 cols each
  const int el   = tid >> 1;
  const int half = tid & 1;
  float sum = 0.f;
#pragma unroll 8
  for (int j = 0; j < 64; ++j) {
    const int col = half * 64 + j;
    sum += hid[el * HHX + col] * sw2[col];
  }
  sum += __shfl_xor(sum, 1, 32);
  if (half == 0) {
    const int e = e0 + el;
    if (e < E) {
      const float eml = sum + b2s[0];
      const float s = 1.0f / (1.0f + __expf(-eml * 0.2f));     // sigmoid(eml / 5)
      float v = s * 1.2f - 0.1f;                               // * (zeta-gamma) + gamma
      v = fminf(fmaxf(v, 0.0f), 1.0f);
      eon[e] = v;
      out_em[(size_t)e * KKX + kExp] = v;
      atomicAdd(&nw[sSrc[el]], v);
      atomicAdd(&nw[sDst[el]], v);
    }
  }
}

// ---------------- node mask + masked features ----------------
__global__ void k_node_mask(const float* __restrict__ nw, const float* __restrict__ x,
                            float* __restrict__ hout, float* __restrict__ out_nm,
                            int kExp, int n) {
  int t = blockIdx.x * blockDim.x + threadIdx.x;
  int node = t >> 5, lane = t & 31;
  if (node >= n) return;
  float w = nw[node] > 0.0f ? 1.0f : 0.0f;
  if (lane == 0) out_nm[(size_t)node * KKX + kExp] = w;
  float4 v = *(const float4*)(x + (size_t)node * HHX + lane * 4);
  v.x *= w; v.y *= w; v.z *= w; v.w *= w;
  *(float4*)(hout + (size_t)node * HHX + lane * 4) = v;
}

// ---------------- pooling helpers ----------------
__global__ void k_count(const int* __restrict__ batch, float* __restrict__ cnt, int n) {
  int i = blockIdx.x * blockDim.x + threadIdx.x;
  if (i < n) atomicAdd(&cnt[batch[i]], 1.0f);
}

__global__ void k_pool_scatter(const float* __restrict__ h, const int* __restrict__ batch,
                               float* __restrict__ pooled, int n) {
  int t = blockIdx.x * blockDim.x + threadIdx.x;
  int node = t >> 5, lane = t & 31;
  if (node >= n) return;
  int g = batch[node];
  const float4 v = *(const float4*)(h + (size_t)node * HHX + lane * 4);
  float* p = pooled + (size_t)g * HHX + lane * 4;
  atomicAdd(p + 0, v.x);
  atomicAdd(p + 1, v.y);
  atomicAdd(p + 2, v.z);
  atomicAdd(p + 3, v.w);
}

__global__ void k_div_store(const float* __restrict__ pooled, const float* __restrict__ cnt,
                            float* __restrict__ dst, int total) {
  int i = blockIdx.x * blockDim.x + threadIdx.x;
  if (i < total) {
    int g = i >> 7;
    dst[i] = pooled[i] / fmaxf(cnt[g], 1.0f);
  }
}

// ---------------- per-graph classifier head ----------------
__launch_bounds__(128)
__global__ void k_head(const float* __restrict__ pooled, const float* __restrict__ cnt,
                       const float* __restrict__ W1, const float* __restrict__ b1,
                       const float* __restrict__ W2, const float* __restrict__ b2,
                       float* __restrict__ out_logits, float* __restrict__ out_h,
                       int kExp) {
  __shared__ float sh[HHX];
  __shared__ float sh2[HHX];
  const int g = blockIdx.x, f = threadIdx.x;
  const float inv = 1.0f / fmaxf(cnt[g], 1.0f);
  const float hv = pooled[(size_t)g * HHX + f] * inv;
  sh[f] = hv;
  out_h[((size_t)g * KKX + kExp) * HHX + f] = hv;
  __syncthreads();
  float a = b1[f];
#pragma unroll 4
  for (int j = 0; j < HHX; ++j) a += sh[j] * W1[j * HHX + f];
  sh2[f] = a > 0.f ? a : 0.f;
  __syncthreads();
  if (f < CCX) {
    float o = b2[f];
#pragma unroll 4
    for (int j = 0; j < HHX; ++j) o += sh2[j] * W2[j * CCX + f];
    out_logits[((size_t)g * KKX + kExp) * CCX + f] = o;
  }
}

// ======================= host side =======================
extern "C" void kernel_launch(void* const* d_in, const int* in_sizes, int n_in,
                              void* d_out, int out_size, void* d_ws, size_t ws_size,
                              hipStream_t stream) {
  const float* x     = (const float*)d_in[0];
  const int*   ei    = (const int*)d_in[1];
  const int*   batch = (const int*)d_in[2];
  const float* ceW1  = (const float*)d_in[3];
  const float* ceb1  = (const float*)d_in[4];
  const float* ceW2  = (const float*)d_in[5];
  const float* ceb2  = (const float*)d_in[6];
  const float* ceeps = (const float*)d_in[7];
  const float* emW1  = (const float*)d_in[8];
  const float* emb1  = (const float*)d_in[9];
  const float* emW2  = (const float*)d_in[10];
  const float* emb2  = (const float*)d_in[11];
  const float* clW1  = (const float*)d_in[12];
  const float* clb1  = (const float*)d_in[13];
  const float* clW2  = (const float*)d_in[14];
  const float* clb2  = (const float*)d_in[15];
  const float* cleps = (const float*)d_in[16];
  const float* ccW1  = (const float*)d_in[17];
  const float* ccb1  = (const float*)d_in[18];
  const float* ccW2  = (const float*)d_in[19];
  const float* ccb2  = (const float*)d_in[20];

  float* out        = (float*)d_out;
  float* out_logits = out;                                  // [G,K,C]
  float* out_hstab  = out + (size_t)GGX * KKX * CCX;        // [G,K,H]
  float* out_horig  = out_hstab + (size_t)GGX * KKX * HHX;  // [G,H]
  float* out_nmask  = out_horig + (size_t)GGX * HHX;        // [N,K]
  float* out_emask  = out_nmask + (size_t)NN * KKX;         // [E,K]

  // workspace carve-up
  char* wp = (char*)d_ws;
  auto carve = [&](size_t bytes) -> void* {
    void* p = (void*)wp;
    wp += (bytes + 255) & ~(size_t)255;
    return p;
  };
  float* hA     = (float*)carve((size_t)NF * 4);
  float* hB     = (float*)carve((size_t)NF * 4);
  float* agg    = (float*)carve((size_t)NF * 4);
  float* Zf     = (float*)carve((size_t)NF * 4);
  bf16t* Zbf    = (bf16t*)carve((size_t)NF * 2);
  float* eon    = (float*)carve((size_t)EE * 4);
  float* nw     = (float*)carve((size_t)NN * 4);
  float* pooled = (float*)carve((size_t)GGX * HHX * 4);
  float* cnt    = (float*)carve((size_t)GGX * 4);
  bf16t* ceW1b  = (bf16t*)carve((size_t)LLX * HHX * HHX * 2);
  bf16t* ceW2b  = (bf16t*)carve((size_t)LLX * HHX * HHX * 2);
  bf16t* clW1b  = (bf16t*)carve((size_t)KKX * LLX * HHX * HHX * 2);
  bf16t* clW2b  = (bf16t*)carve((size_t)KKX * LLX * HHX * HHX * 2);
  bf16t* emW1b  = (bf16t*)carve((size_t)KKX * 2 * HHX * HHX * 2);

  // convert + transpose weights to bf16 [n][k]
  auto convT = [&](const float* s, bf16t* d, int K, int N, int mats) {
    int total = mats * K * N;
    k_convT_bf16<<<(total + 255) / 256, 256, 0, stream>>>(s, d, K, N, total);
  };
  convT(ceW1, ceW1b, HHX, HHX, LLX);
  convT(ceW2, ceW2b, HHX, HHX, LLX);
  convT(clW1, clW1b, HHX, HHX, KKX * LLX);
  convT(clW2, clW2b, HHX, HHX, KKX * LLX);
  convT(emW1, emW1b, 2 * HHX, HHX, KKX);

  // graph sizes
  hipMemsetAsync(cnt, 0, (size_t)GGX * 4, stream);
  k_count<<<(NN + 255) / 256, 256, 0, stream>>>(batch, cnt, NN);

  const int gsGrid = ((EE * 32) + 255) / 256;
  const int ginGrid = (NN + 63) / 64;

  // ---- causal encoder: 3 GIN layers ----
  const float* hin = x;
  float* encOut[3] = {hA, hB, Zf};
  for (int l = 0; l < LLX; ++l) {
    hipMemsetAsync(agg, 0, (size_t)NF * 4, stream);
    k_gather_scatter<<<gsGrid, 256, 0, stream>>>(hin, ei, nullptr, agg, EE);
    k_gin_update_wmma<<<ginGrid, 128, 0, stream>>>(
        hin, agg, ceeps, l,
        ceW1b + (size_t)l * HHX * HHX, ceb1 + l * HHX,
        ceW2b + (size_t)l * HHX * HHX, ceb2 + l * HHX,
        encOut[l], (l == LLX - 1) ? Zbf : (bf16t*)nullptr, NN);
    hin = encOut[l];
  }

  // ---- h_orig = mean_pool(Z) ----
  hipMemsetAsync(pooled, 0, (size_t)GGX * HHX * 4, stream);
  k_pool_scatter<<<((NN * 32) + 255) / 256, 256, 0, stream>>>(Zf, batch, pooled, NN);
  k_div_store<<<((GGX * HHX) + 255) / 256, 256, 0, stream>>>(pooled, cnt, out_horig, GGX * HHX);

  // ---- experts ----
  for (int k = 0; k < KKX; ++k) {
    hipMemsetAsync(nw, 0, (size_t)NN * 4, stream);
    k_edge_mask_wmma<<<(EE + 63) / 64, 128, 0, stream>>>(
        Zbf, ei,
        emW1b + (size_t)k * 2 * HHX * HHX, emb1 + k * HHX,
        emW2 + k * HHX, emb2 + k,
        eon, nw, out_emask, k, EE);
    k_node_mask<<<((NN * 32) + 255) / 256, 256, 0, stream>>>(nw, x, hA, out_nmask, k, NN);

    const float* hcur = hA;
    for (int l = 0; l < LLX; ++l) {
      hipMemsetAsync(agg, 0, (size_t)NF * 4, stream);
      k_gather_scatter<<<gsGrid, 256, 0, stream>>>(hcur, ei, eon, agg, EE);
      float* ho = (l == 1) ? hA : hB;
      const size_t wOff = (size_t)(k * LLX + l) * HHX * HHX;
      k_gin_update_wmma<<<ginGrid, 128, 0, stream>>>(
          hcur, agg, cleps, k * LLX + l,
          clW1b + wOff, clb1 + (size_t)(k * LLX + l) * HHX,
          clW2b + wOff, clb2 + (size_t)(k * LLX + l) * HHX,
          ho, (bf16t*)nullptr, NN);
      hcur = ho;
    }

    hipMemsetAsync(pooled, 0, (size_t)GGX * HHX * 4, stream);
    k_pool_scatter<<<((NN * 32) + 255) / 256, 256, 0, stream>>>(hcur, batch, pooled, NN);
    k_head<<<GGX, 128, 0, stream>>>(
        pooled, cnt,
        ccW1 + (size_t)k * HHX * HHX, ccb1 + (size_t)k * HHX,
        ccW2 + (size_t)k * HHX * CCX, ccb2 + (size_t)k * CCX,
        out_logits, out_hstab, k);
  }
}
